// Linear_Transformer_Decoder_45037027066199
// MI455X (gfx1250) — matrix operand order repaired and checked by reading, not compiled, not run on hardware
//
#include <hip/hip_runtime.h>

typedef __attribute__((ext_vector_type(16))) _Float16 v16h;
typedef __attribute__((ext_vector_type(8)))  _Float16 v8h;
typedef __attribute__((ext_vector_type(4)))  _Float16 v4h;
typedef __attribute__((ext_vector_type(8)))  float    v8f;
typedef __attribute__((ext_vector_type(4)))  unsigned int u32x4;
typedef __attribute__((ext_vector_type(8)))  int      i32x8;
typedef __attribute__((ext_vector_type(4)))  int      i32x4;

#define L_SEQ  4096
#define BATCH  8
#define DIM    256
#define HEADS  8
#define DH     32
#define FFN    1024
#define ROWS   (BATCH * L_SEQ)   // 32768 tokens
#define KCHUNK 256               // K panel resident in LDS per TDM transfer

#if defined(__HIP_DEVICE_COMPILE__) && __has_builtin(__builtin_amdgcn_tensor_load_to_lds) && __has_builtin(__builtin_amdgcn_s_wait_tensorcnt)
#define USE_TDM 1
#else
#define USE_TDM 0
#endif

// ---------------------------------------------------------------------------
// WMMA fragment loaders (wave32 layouts, CDNA5 ISA §7.12.2)
// A (16x32 f16): lane m (0-15) = row m; half h = lane>>4:
//   elems 0..7 -> K = h*8+0..7 ; elems 8..15 -> K = 16+h*8+0..7
// B (32x16 f16): lane n = column n; same K order as A (h*8+0..7, 16+h*8+0..7).
// ---------------------------------------------------------------------------
__device__ __forceinline__ v16h frag_a(const _Float16* rowbase, int h) {
  v8h lo = *(const v8h*)(rowbase + h * 8);
  v8h hi = *(const v8h*)(rowbase + 16 + h * 8);
  return __builtin_shufflevector(lo, hi, 0,1,2,3,4,5,6,7,8,9,10,11,12,13,14,15);
}
__device__ __forceinline__ v16h frag_b(const _Float16* colbase, int h) {
  v8h lo = *(const v8h*)(colbase + h * 8);
  v8h hi = *(const v8h*)(colbase + 16 + h * 8);
  return __builtin_shufflevector(lo, hi, 0,1,2,3,4,5,6,7,8,9,10,11,12,13,14,15);
}
__device__ __forceinline__ v8f wmma_f16(v16h a, v16h b, v8f c) {
  return __builtin_amdgcn_wmma_f32_16x16x32_f16(false, a, false, b, (short)0, c, false, false);
}

#if USE_TDM
// ---------------------------------------------------------------------------
// TDM 2D tile load: global (row-major, stride s0 elems) -> LDS (packed tile).
// Descriptor per CDNA5 ISA:
//   group0: count=1 | lds_addr | global_addr[56:0] | type=2
//   group1: data_size=1(2B) | tensor_dim0/1 | tile_dim0/1 | dim0_stride
// 2D tensor: remaining groups zero. Tracked by TENSORcnt.
// This toolchain's builtin takes 6 args (g0, g1, g2, g3, g4, cpol).
// ---------------------------------------------------------------------------
__device__ __forceinline__ void tdm_load_2d(unsigned lds_off, const void* gptr,
                                            unsigned td0, unsigned td1,
                                            unsigned tl0, unsigned tl1,
                                            unsigned long long s0) {
  const unsigned long long ga = (unsigned long long)gptr;
  u32x4 g0;
  g0[0] = 1u;                                            // count=1, user mode
  g0[1] = lds_off;                                       // LDS byte address
  g0[2] = (unsigned)ga;                                  // global_addr[31:0]
  g0[3] = (unsigned)((ga >> 32) & 0x1FFFFFFu) | (2u << 30);  // [56:32] | type=2
  i32x8 g1;
  g1[0] = (int)(1u << 16);                               // data_size = 2 bytes
  g1[1] = (int)((td0 & 0xFFFFu) << 16);                  // tensor_dim0[15:0]
  g1[2] = (int)((td0 >> 16) | ((td1 & 0xFFFFu) << 16));  // dim0 hi | dim1 lo
  g1[3] = (int)((td1 >> 16) | (tl0 << 16));              // dim1 hi | tile_dim0
  g1[4] = (int)(tl1 & 0xFFFFu);                          // tile_dim1 (tile_dim2=0)
  g1[5] = (int)(unsigned)s0;                             // dim0_stride[31:0]
  g1[6] = (int)(unsigned)((s0 >> 32) & 0xFFFFu);         // dim0_stride[47:32]
  g1[7] = 0;
  const i32x4 z4 = {0, 0, 0, 0};
  const i32x8 z8 = {0, 0, 0, 0, 0, 0, 0, 0};
  __builtin_amdgcn_tensor_load_to_lds(g0, g1, z4, z4, z8, 0);
}
// ISA §10.2: flat LDS-aperture address low 32 bits are the LDS byte offset.
__device__ __forceinline__ unsigned lds_off_of(const void* p) {
  return (unsigned)(unsigned long long)p;
}
#endif

// ---------------------------------------------------------------------------
// GEMM: C[M,N](f16) = act(A[M,K](f16) * W + bias), W given transposed: WT[N,K]
// ACT: 0 none, 1 relu, 2 elu(x)+1.
// 256 thr = 8 waves; block tile 128x64; wave tile 32x32.
// TDM path: per 256-K chunk one tensor_load pair fills 96KB LDS panels, then
// 8 barrier-free WMMA sub-steps (32 WMMA between barriers, no staging code).
// Fallback: round-2 double-buffered register staging (BK=32).
// ---------------------------------------------------------------------------
template <int ACT>
__global__ __launch_bounds__(256)
void gemm_f16_kernel(const _Float16* __restrict__ A, const _Float16* __restrict__ WT,
                     const float* __restrict__ bias, _Float16* __restrict__ C,
                     int M, int K, int N) {
  extern __shared__ __align__(16) _Float16 smem[];
  const int t = threadIdx.x;
  const int w = t >> 5, lane = t & 31, l15 = lane & 15, hh = lane >> 4;
  const int wm = w & 3, wn = w >> 2;                 // 4 x 2 wave grid
  const int m0 = blockIdx.y * 128, n0 = blockIdx.x * 64;
  v8f acc00 = {}, acc01 = {}, acc10 = {}, acc11 = {};

#if USE_TDM
  _Float16* lA = smem;                   // [128][KCHUNK]
  _Float16* lB = smem + 128 * KCHUNK;    // [64][KCHUNK]
  for (int kc = 0; kc < K; kc += KCHUNK) {
    if (t < 32) {                        // one wave programs the TDM
      tdm_load_2d(lds_off_of(lA), A  + (size_t)m0 * K + kc,
                  (unsigned)K, (unsigned)M, KCHUNK, 128, (unsigned long long)K);
      tdm_load_2d(lds_off_of(lB), WT + (size_t)n0 * K + kc,
                  (unsigned)K, (unsigned)N, KCHUNK, 64, (unsigned long long)K);
      __builtin_amdgcn_s_wait_tensorcnt(0);
    }
    __syncthreads();
#pragma unroll
    for (int k0 = 0; k0 < KCHUNK; k0 += 32) {
      v16h a0 = frag_a(&lA[(wm * 32 + l15) * KCHUNK + k0], hh);
      v16h a1 = frag_a(&lA[(wm * 32 + 16 + l15) * KCHUNK + k0], hh);
      v16h b0 = frag_b(&lB[(wn * 32 + l15) * KCHUNK + k0], hh);
      v16h b1 = frag_b(&lB[(wn * 32 + 16 + l15) * KCHUNK + k0], hh);
      acc00 = wmma_f16(a0, b0, acc00);
      acc01 = wmma_f16(a0, b1, acc01);
      acc10 = wmma_f16(a1, b0, acc10);
      acc11 = wmma_f16(a1, b1, acc11);
    }
    __syncthreads();
  }
#else
  _Float16* lA = smem;                   // [2][128*32]
  _Float16* lB = smem + 2 * 128 * 32;    // [2][64*32]
  const int ar = t >> 2, ac = (t & 3) * 8;
  const int br = t >> 2, bc = (t & 3) * 8;
  const _Float16* pa0 = A  + (size_t)(m0 + ar) * K + ac;
  const _Float16* pa1 = A  + (size_t)(m0 + ar + 64) * K + ac;
  const _Float16* pb  = WT + (size_t)(n0 + br) * K + bc;
  uint4 ra0 = *(const uint4*)pa0;
  uint4 ra1 = *(const uint4*)pa1;
  uint4 rb0 = *(const uint4*)pb;
  int buf = 0;
  for (int k0 = 0; k0 < K; k0 += 32) {
    *(uint4*)&lA[buf * 128 * 32 + ar * 32 + ac]        = ra0;
    *(uint4*)&lA[buf * 128 * 32 + (ar + 64) * 32 + ac] = ra1;
    *(uint4*)&lB[buf * 64 * 32 + br * 32 + bc]         = rb0;
    __syncthreads();
    if (k0 + 32 < K) {
      ra0 = *(const uint4*)(pa0 + k0 + 32);
      ra1 = *(const uint4*)(pa1 + k0 + 32);
      rb0 = *(const uint4*)(pb  + k0 + 32);
    }
    v16h a0 = frag_a(&lA[buf * 128 * 32 + (wm * 32 + l15) * 32], hh);
    v16h a1 = frag_a(&lA[buf * 128 * 32 + (wm * 32 + 16 + l15) * 32], hh);
    v16h b0 = frag_b(&lB[buf * 64 * 32 + (wn * 32 + l15) * 32], hh);
    v16h b1 = frag_b(&lB[buf * 64 * 32 + (wn * 32 + 16 + l15) * 32], hh);
    acc00 = wmma_f16(a0, b0, acc00);
    acc01 = wmma_f16(a0, b1, acc01);
    acc10 = wmma_f16(a1, b0, acc10);
    acc11 = wmma_f16(a1, b1, acc11);
    buf ^= 1;
    __syncthreads();
  }
#endif

  const v8f* accs[4] = { &acc00, &acc01, &acc10, &acc11 };
#pragma unroll
  for (int i = 0; i < 2; ++i) {
#pragma unroll
    for (int j = 0; j < 2; ++j) {
      const v8f& a = *accs[i * 2 + j];
#pragma unroll
      for (int r = 0; r < 8; ++r) {
        const int m = m0 + wm * 32 + i * 16 + r + 8 * hh;
        const int n = n0 + wn * 32 + j * 16 + l15;
        float v = a[r] + bias[n];
        if (ACT == 1) v = fmaxf(v, 0.f);
        if (ACT == 2) v = v > 0.f ? v + 1.f : __expf(v);
        C[(size_t)m * N + n] = (_Float16)v;
      }
    }
  }
}

// ---------------------------------------------------------------------------
// KV[b,h] (32x32) += Kf_chunk^T @ V_chunk ; Ksum[b,h] += colsum(Kf_chunk)
// grid: x = 4 seq chunks of 1024, y = 64 (b,h). Waves 0-3 WMMA, wave 4 Ksum.
// f32 accumulation into global via global_atomic_add_f32.
// ---------------------------------------------------------------------------
__global__ __launch_bounds__(256)
void kv_reduce_kernel(const _Float16* __restrict__ kb, const _Float16* __restrict__ vb,
                      float* __restrict__ KV, float* __restrict__ Ksum) {
  __shared__ __align__(16) _Float16 lKT[32 * 32];
  __shared__ __align__(16) _Float16 lVT[32 * 32];
  const int bh = blockIdx.y, b = bh >> 3, hd = bh & 7;
  const int t = threadIdx.x, w = t >> 5, lane = t & 31, l15 = lane & 15, hh = lane >> 4;
  const int sl = t >> 3, d0 = (t & 7) * 4;
  const size_t baserow = (size_t)b * L_SEQ + (size_t)blockIdx.x * 1024;
  const int dt = w & 1, mt = w >> 1;
  v8f acc = {};
  float ksl = 0.f;

  for (int l0 = 0; l0 < 1024; l0 += 32) {
    const v4h k4 = *(const v4h*)(kb + (baserow + l0 + sl) * DIM + hd * DH + d0);
    const v4h v4 = *(const v4h*)(vb + (baserow + l0 + sl) * DIM + hd * DH + d0);
#pragma unroll
    for (int j = 0; j < 4; ++j) {
      lKT[(d0 + j) * 32 + sl] = k4[j];
      lVT[(d0 + j) * 32 + sl] = v4[j];
    }
    __syncthreads();
    if (w < 4) {
      v16h af = frag_a(&lKT[(dt * 16 + l15) * 32], hh);
      v16h bf = frag_b(&lVT[(mt * 16 + l15) * 32], hh);
      acc = wmma_f16(af, bf, acc);
    } else if (w == 4) {
#pragma unroll
      for (int j = 0; j < 32; ++j) ksl += (float)lKT[lane * 32 + j];
    }
    __syncthreads();
  }

  if (w < 4) {
#pragma unroll
    for (int r = 0; r < 8; ++r) {
      const int d = dt * 16 + r + 8 * hh, m = mt * 16 + l15;
      unsafeAtomicAdd(&KV[(size_t)bh * (DH * DH) + d * DH + m], acc[r]);
    }
  } else if (w == 4) {
    unsafeAtomicAdd(&Ksum[bh * DH + lane], ksl);
  }
}

// Z[b,l,h] = 1 / (dot(Qf[b,l,h,:], Ksum[b,h,:]) + 1e-6)
__global__ void z_kernel(const _Float16* __restrict__ qb, const float* __restrict__ Ksum,
                         float* __restrict__ Z) {
  const int t = blockIdx.x * blockDim.x + threadIdx.x;
  const int row = t >> 3, hd = t & 7;
  const int b = row / L_SEQ;
  const _Float16* qp = qb + (size_t)row * DIM + hd * DH;
  const float* kp = Ksum + (b * HEADS + hd) * DH;
  float s = 0.f;
#pragma unroll
  for (int d = 0; d < DH; ++d) s += (float)qp[d] * kp[d];
  Z[t] = 1.f / (s + 1e-6f);
}

// out[b,l,h,:] = Z[b,l,h] * (Qf[b,l,h,:] @ KV[b,h])  -- M=4096,K=32,N=32 GEMM
__global__ __launch_bounds__(256)
void attn_apply_kernel(const _Float16* __restrict__ qb, const float* __restrict__ KV,
                       const float* __restrict__ Z, _Float16* __restrict__ out) {
  __shared__ __align__(16) _Float16 lKVT[32 * 32];
  const int bh = blockIdx.y, b = bh >> 3, hd = bh & 7;
  const int t = threadIdx.x, w = t >> 5, lane = t & 31, l15 = lane & 15, hh = lane >> 4;

  {
    const int d = t >> 3, m0 = (t & 7) * 4;
    const float4 kv4 = *(const float4*)(KV + (size_t)bh * (DH * DH) + d * DH + m0);
    lKVT[(m0 + 0) * 32 + d] = (_Float16)kv4.x;
    lKVT[(m0 + 1) * 32 + d] = (_Float16)kv4.y;
    lKVT[(m0 + 2) * 32 + d] = (_Float16)kv4.z;
    lKVT[(m0 + 3) * 32 + d] = (_Float16)kv4.w;
  }
  __syncthreads();

  const size_t rowg = (size_t)b * L_SEQ + (size_t)blockIdx.x * 128 + w * 16 + l15;
  v16h af = frag_a(qb + rowg * DIM + hd * DH, hh);   // direct-from-global A frag
  v16h b0 = frag_b(&lKVT[l15 * 32], hh);
  v16h b1 = frag_b(&lKVT[(16 + l15) * 32], hh);
  v8f a0 = {}, a1 = {};
  a0 = wmma_f16(af, b0, a0);
  a1 = wmma_f16(af, b1, a1);

#pragma unroll
  for (int r = 0; r < 8; ++r) {
    const size_t grow = (size_t)b * L_SEQ + (size_t)blockIdx.x * 128 + w * 16 + r + 8 * hh;
    const float zv = Z[grow * HEADS + hd];
    out[grow * DIM + hd * DH + l15]      = (_Float16)(a0[r] * zv);
    out[grow * DIM + hd * DH + 16 + l15] = (_Float16)(a1[r] * zv);
  }
}

// o[row] = LayerNorm(a[row] + r[row]) * g + be   (wave per 256-wide row)
template <typename TO>
__global__ __launch_bounds__(256)
void addln_kernel(const _Float16* __restrict__ a, const _Float16* __restrict__ r,
                  const float* __restrict__ g, const float* __restrict__ be,
                  TO* __restrict__ o) {
  const int t = threadIdx.x, w = t >> 5, lane = t & 31;
  const size_t row = (size_t)blockIdx.x * 8 + w;
  const _Float16* ap = a + row * DIM;
  const _Float16* rp = r + row * DIM;
  float x[8], s = 0.f;
#pragma unroll
  for (int j = 0; j < 8; ++j) {
    x[j] = (float)ap[lane * 8 + j] + (float)rp[lane * 8 + j];
    s += x[j];
  }
#pragma unroll
  for (int off = 16; off; off >>= 1) s += __shfl_xor(s, off, 32);
  const float mu = s * (1.f / 256.f);
  float vs = 0.f;
#pragma unroll
  for (int j = 0; j < 8; ++j) { const float d = x[j] - mu; vs += d * d; }
#pragma unroll
  for (int off = 16; off; off >>= 1) vs += __shfl_xor(vs, off, 32);
  const float rstd = rsqrtf(vs * (1.f / 256.f) + 1e-3f);
  TO* op = o + row * DIM;
#pragma unroll
  for (int j = 0; j < 8; ++j) {
    const int c = lane * 8 + j;
    op[c] = (TO)((x[j] - mu) * rstd * g[c] + be[c]);
  }
}

// x1 = f16(Q + sine positional encoding)
__global__ void pe_kernel(const float* __restrict__ Q, _Float16* __restrict__ x1) {
  const size_t t = (size_t)blockIdx.x * blockDim.x + threadIdx.x;
  const int d = (int)(t & 255);
  const int l = (int)((t >> 8) & (L_SEQ - 1));
  const float rate = __powf(1e4f, -(float)(2 * (d >> 1)) / 256.f);
  const float ang = (float)l * rate;
  x1[t] = (_Float16)(Q[t] + ((d & 1) ? __cosf(ang) : __sinf(ang)));
}

// f32 -> f16 elementwise copy
__global__ void f2h_kernel(const float* __restrict__ in, _Float16* __restrict__ out) {
  const size_t t = (size_t)blockIdx.x * blockDim.x + threadIdx.x;
  out[t] = (_Float16)in[t];
}

// Weights: W[L,K,N] f32 -> WT[L,N,K] f16 (convert + transpose, once per launch)
__global__ void wt_convert_kernel(const float* __restrict__ W, _Float16* __restrict__ WT,
                                  int K, int N) {
  const size_t t = (size_t)blockIdx.x * blockDim.x + threadIdx.x;   // over L*K*N
  const size_t kn = t % ((size_t)K * N);
  const size_t l = t / ((size_t)K * N);
  const int k = (int)(kn / N), n = (int)(kn % N);
  WT[l * (size_t)K * N + (size_t)n * K + k] = (_Float16)W[t];
}

// ---------------------------------------------------------------------------
extern "C" void kernel_launch(void* const* d_in, const int* in_sizes, int n_in,
                              void* d_out, int out_size, void* d_ws, size_t ws_size,
                              hipStream_t stream) {
  (void)in_sizes; (void)n_in; (void)out_size; (void)ws_size;
  const float* Qin = (const float*)d_in[0];
  const float* Kin = (const float*)d_in[1];
  const float* Wf32[10] = { (const float*)d_in[2],  (const float*)d_in[3],
                            (const float*)d_in[4],  (const float*)d_in[5],
                            (const float*)d_in[10], (const float*)d_in[11],
                            (const float*)d_in[12], (const float*)d_in[13],
                            (const float*)d_in[18], (const float*)d_in[20] };
  const float* bq1 = (const float*)d_in[6];
  const float* bk1 = (const float*)d_in[7];
  const float* bv1 = (const float*)d_in[8];
  const float* bo1 = (const float*)d_in[9];
  const float* bq2 = (const float*)d_in[14];
  const float* bk2 = (const float*)d_in[15];
  const float* bv2 = (const float*)d_in[16];
  const float* bo2 = (const float*)d_in[17];
  const float* bf1 = (const float*)d_in[19];
  const float* bf2 = (const float*)d_in[21];
  const float* ln1g = (const float*)d_in[22];
  const float* ln1b = (const float*)d_in[23];
  const float* ln2g = (const float*)d_in[24];
  const float* ln2b = (const float*)d_in[25];
  const float* ln3g = (const float*)d_in[26];
  const float* ln3b = (const float*)d_in[27];

  // ---- workspace carve (f16 activations, f32 accumulators) ----
  char* p = (char*)d_ws;
  auto allocH = [&](size_t n) { _Float16* r = (_Float16*)p; p += n * sizeof(_Float16); return r; };
  const size_t SZH = (size_t)ROWS * DIM;
  _Float16* x1 = allocH(SZH);
  _Float16* x2 = allocH(SZH);
  _Float16* x3 = allocH(SZH);
  _Float16* qb = allocH(SZH);
  _Float16* kb = allocH(SZH);
  _Float16* vb = allocH(SZH);
  _Float16* at = allocH(SZH);
  _Float16* pj = allocH(SZH);
  _Float16* hb = allocH((size_t)ROWS * FFN);
  _Float16* kin16 = allocH(SZH);
  _Float16* WT[10];
  for (int i = 0; i < 10; ++i)
    WT[i] = allocH((size_t)6 * (i < 8 ? DIM * DIM : DIM * FFN));
  float* kv = (float*)p;
  float* ks = kv + 64 * DH * DH;
  float* zb = ks + 64 * DH;

  const dim3 blk(256);
  const size_t gemm_lds = (size_t)(128 + 64) * KCHUNK * sizeof(_Float16);  // 96 KB

  // one-time conversions: weights (convert+transpose), inputs -> f16
  for (int i = 0; i < 8; ++i)
    wt_convert_kernel<<<(6 * DIM * DIM) / 256, blk, 0, stream>>>(Wf32[i], WT[i], DIM, DIM);
  wt_convert_kernel<<<(6 * DIM * FFN) / 256, blk, 0, stream>>>(Wf32[8], WT[8], DIM, FFN);
  wt_convert_kernel<<<(6 * FFN * DIM) / 256, blk, 0, stream>>>(Wf32[9], WT[9], FFN, DIM);
  f2h_kernel<<<(ROWS * DIM) / 256, blk, 0, stream>>>(Kin, kin16);
  pe_kernel<<<(ROWS * DIM) / 256, blk, 0, stream>>>(Qin, x1);

  auto gemm = [&](int act, const _Float16* A, const _Float16* Wt, const float* bias,
                  _Float16* C, int M, int K, int N) {
    dim3 g(N / 64, M / 128);
    if (act == 0)      gemm_f16_kernel<0><<<g, blk, gemm_lds, stream>>>(A, Wt, bias, C, M, K, N);
    else if (act == 1) gemm_f16_kernel<1><<<g, blk, gemm_lds, stream>>>(A, Wt, bias, C, M, K, N);
    else               gemm_f16_kernel<2><<<g, blk, gemm_lds, stream>>>(A, Wt, bias, C, M, K, N);
  };
  auto lin_attn = [&](const _Float16* xq, const _Float16* xkv,
                      const _Float16* wq, const float* bq, const _Float16* wk, const float* bk_,
                      const _Float16* wv, const float* bv, const _Float16* wo, const float* bo) {
    gemm(2, xq,  wq, bq,  qb, ROWS, DIM, DIM);   // Qf = elu(q)+1
    gemm(2, xkv, wk, bk_, kb, ROWS, DIM, DIM);   // Kf = elu(k)+1
    gemm(0, xkv, wv, bv,  vb, ROWS, DIM, DIM);   // v
    (void)hipMemsetAsync(kv, 0, (size_t)(64 * DH * DH + 64 * DH) * sizeof(float), stream);
    kv_reduce_kernel<<<dim3(4, 64), blk, 0, stream>>>(kb, vb, kv, ks);
    z_kernel<<<(ROWS * HEADS) / 256, blk, 0, stream>>>(qb, ks, zb);
    attn_apply_kernel<<<dim3(L_SEQ / 128, 64), blk, 0, stream>>>(qb, kv, zb, at);
    gemm(0, at, wo, bo, pj, ROWS, DIM, DIM);     // output projection
  };

  for (int i = 0; i < 6; ++i) {
    const size_t wo_ = (size_t)i * DIM * DIM, bo_ = (size_t)i * DIM;
    // self attention on x1
    lin_attn(x1, x1, WT[0] + wo_, bq1 + bo_, WT[1] + wo_, bk1 + bo_,
             WT[2] + wo_, bv1 + bo_, WT[3] + wo_, bo1 + bo_);
    addln_kernel<_Float16><<<ROWS / 8, blk, 0, stream>>>(pj, x1, ln1g + bo_, ln1b + bo_, x2);
    // cross attention: queries x2, keys/values from encoder input K
    lin_attn(x2, kin16, WT[4] + wo_, bq2 + bo_, WT[5] + wo_, bk2 + bo_,
             WT[6] + wo_, bv2 + bo_, WT[7] + wo_, bo2 + bo_);
    addln_kernel<_Float16><<<ROWS / 8, blk, 0, stream>>>(pj, x2, ln2g + bo_, ln2b + bo_, x3);
    // FFN (both denses have relu per reference)
    gemm(1, x3, WT[8] + (size_t)i * DIM * FFN, bf1 + (size_t)i * FFN, hb, ROWS, DIM, FFN);
    gemm(1, hb, WT[9] + (size_t)i * FFN * DIM, bf2 + bo_, pj, ROWS, FFN, DIM);
    if (i == 5)
      addln_kernel<float><<<ROWS / 8, blk, 0, stream>>>(pj, x3, ln3g + bo_, ln3b + bo_, (float*)d_out);
    else
      addln_kernel<_Float16><<<ROWS / 8, blk, 0, stream>>>(pj, x3, ln3g + bo_, ln3b + bo_, x1);
  }
}
